// SiamFCDualPathXCorr_80942953661001
// MI455X (gfx1250) — matrix operand order repaired and checked by reading, not compiled, not run on hardware
//
#include <hip/hip_runtime.h>
#include <hip/hip_bf16.h>

typedef __attribute__((ext_vector_type(16))) _Float16 v16h;
typedef __attribute__((ext_vector_type(8)))  _Float16 v8h;
typedef __attribute__((ext_vector_type(8)))  float    v8f;
typedef __attribute__((ext_vector_type(4)))  int      v4i;

#define CIN   256
#define COUT  256
#define KTOT  2304   // 9 taps * 256 cin, k = t*256 + cin
#define NB    32     // batch

#if __has_builtin(__builtin_amdgcn_global_load_async_to_lds_b128) && \
    __has_builtin(__builtin_amdgcn_s_wait_asynccnt)
#define USE_ASYNC 1
typedef __attribute__((address_space(1))) v4i gv4i;   // global v4i
typedef __attribute__((address_space(3))) v4i lv4i;   // LDS v4i
#else
#define USE_ASYNC 0
#endif

// ---------------------------------------------------------------------------
// Fold BN into weights, convert to f16, re-layout [cout][t][cin].
// grid = 2304 blocks * 256 threads = 256*2304 elements
// ---------------------------------------------------------------------------
__global__ __launch_bounds__(256) void prep_weights_kernel(
    const float* __restrict__ w, const float* __restrict__ g,
    const float* __restrict__ bb, const float* __restrict__ m,
    const float* __restrict__ v, _Float16* __restrict__ wout,
    float* __restrict__ beta)
{
    int idx  = blockIdx.x * 256 + threadIdx.x;      // [0, 256*2304)
    int cout = idx / KTOT;
    int k    = idx % KTOT;
    int t    = k / CIN;
    int cin  = k % CIN;
    int ky = t / 3, kx = t % 3;
    float inv = g[cout] * rsqrtf(v[cout] + 1e-5f);
    wout[(size_t)cout * KTOT + k] =
        (_Float16)(w[((cout * CIN + cin) * 3 + ky) * 3 + kx] * inv);
    if (k == 0) beta[cout] = bb[cout] - m[cout] * inv;
}

// ---------------------------------------------------------------------------
// Convert NCHW f32 -> channel-last f16 [b][pos][cin] via LDS tile transpose.
// Block: 256 threads, tile = 32 cin x 32 pos. grid = NB * 8 * ceil(HW/32)
// ---------------------------------------------------------------------------
__global__ __launch_bounds__(256) void convert_nhwc_kernel(
    const float* __restrict__ X, _Float16* __restrict__ Xt, int HW)
{
    __shared__ float t[32][33];
    const int nPos = (HW + 31) >> 5;
    const int blk = blockIdx.x;
    const int pt = blk % nPos;
    const int ct = (blk / nPos) & 7;
    const int b  = blk / (nPos * 8);
    const int tx = threadIdx.x & 31;
    const int ty = threadIdx.x >> 5;        // 32x8 thread tile

#pragma unroll
    for (int gI = 0; gI < 4; ++gI) {
        int cin = ct * 32 + ty + gI * 8;
        int pos = pt * 32 + tx;
        float vv = 0.f;
        if (pos < HW) vv = X[((size_t)b * CIN + cin) * HW + pos];
        t[ty + gI * 8][tx] = vv;            // t[cin_local][pos_local]
    }
    __syncthreads();
#pragma unroll
    for (int gI = 0; gI < 4; ++gI) {
        int posl = ty + gI * 8;
        int pos = pt * 32 + posl;
        if (pos < HW)
            Xt[((size_t)b * HW + pos) * CIN + ct * 32 + tx] = (_Float16)t[tx][posl];
    }
}

// ---------------------------------------------------------------------------
// Implicit-GEMM 3x3 VALID conv + folded BN, f16 WMMA, f32 accumulate.
// M = cout (256), N = Wout*Wout positions, K = 2304.
// Block: 256 threads = 8 waves; tile 64M x 128N; wave tile 32x32 (4 wmma acc).
// Activations are channel-last f16 so the LDS slab [pc][cin32] is staged with
// async-to-LDS 16B copies, double-buffered across cin chunks. A (weight)
// fragments are software-pipelined one tap ahead to hide L2 latency.
// grid.x = NB * 4 * nTilesN
// ---------------------------------------------------------------------------
template <int HIN>
__global__ __launch_bounds__(256) void conv_gemm_kernel(
    const _Float16* __restrict__ Xt,    // [NB, HIN*HIN, CIN] channel-last f16
    const _Float16* __restrict__ Wp,    // [COUT][KTOT] packed f16
    const float*    __restrict__ beta,  // [COUT]
    _Float16*       __restrict__ Y,     // [NB, COUT, P]
    int Wout, int P, int nTilesN)
{
    constexpr int NPT = (32 * HIN) / 256;        // 16B async transfers per thread
    __shared__ _Float16 slab[2][8 * HIN * 32];   // [buf][pc=row*HIN+col][cin32]

    const int blk = blockIdx.x;
    const int nt  = blk % nTilesN;
    const int mt  = (blk / nTilesN) & 3;
    const int b   = blk / (nTilesN * 4);

    const int tid   = threadIdx.x;
    const int lane  = tid & 31;
    const int wid   = tid >> 5;
    const int waveM = wid & 1;        // 2 -> 64 M per block
    const int waveN = wid >> 1;       // 4 -> 128 N per block
    const int l15   = lane & 15;
    const int hi    = lane >> 4;

    const int p0  = nt * 128;
    const int oy0 = p0 / Wout;

    // Stage cin-chunk cc into slab[buf]: 32*HIN 16-byte transfers.
    auto stage = [&](int buf, int cc) {
#pragma unroll
        for (int j = 0; j < NPT; ++j) {
            int i   = tid + j * 256;
            int seg = i & 3;                  // 8-cin segment within chunk
            int pc  = i >> 2;                 // r*HIN + col
            int lin = oy0 * HIN + pc;         // absolute row*HIN + col
            if (lin > HIN * HIN - 1) lin = HIN * HIN - 1;   // stay in-bounds (tail)
            const _Float16* gp = Xt + ((size_t)b * HIN * HIN + lin) * CIN + cc * 32 + seg * 8;
            _Float16* lp = &slab[buf][pc * 32 + seg * 8];
#if USE_ASYNC
            __builtin_amdgcn_global_load_async_to_lds_b128(
                (gv4i*)(void*)gp, (lv4i*)(void*)lp, 0, 0);
#else
            *(v8h*)lp = *(const v8h*)gp;
#endif
        }
    };

    // Per-lane B-fragment positions (fixed for whole kernel). C/D: N = lane&15.
    int pos[2], py[2], px[2];
#pragma unroll
    for (int nf = 0; nf < 2; ++nf) {
        int p = p0 + waveN * 32 + nf * 16 + l15;
        pos[nf] = p;
        int pc = (p < P) ? p : (P - 1);
        py[nf] = pc / Wout;
        px[nf] = pc % Wout;
    }

    // A-fragment row base pointers. 16-bit A: lanes0-15 K=0..7,16..23; lanes16-31 +8.
    const _Float16* arow[2];
#pragma unroll
    for (int mf = 0; mf < 2; ++mf) {
        int row = mt * 64 + waveM * 32 + mf * 16 + l15;
        arow[mf] = Wp + (size_t)row * KTOT + hi * 8;
    }

    // Load both A fragments for (chunk cc, tap t).
    auto loadA = [&](int cc, int t, v16h af[2]) {
        const int kbase = t * CIN + cc * 32;
#pragma unroll
        for (int mf = 0; mf < 2; ++mf) {
            union { v16h v; v8h h[2]; } u;
            u.h[0] = *(const v8h*)(arow[mf] + kbase);        // K 0..7 (+hi*8)
            u.h[1] = *(const v8h*)(arow[mf] + kbase + 16);   // K 16..23 (+hi*8)
            af[mf] = u.v;
        }
    };

    v8f acc[2][2] = {};
    v16h acur[2], anxt[2];

    stage(0, 0);                                  // prologue: chunk 0 DMA in flight
    loadA(0, 0, acur);                            // prologue: tap-0 weights in flight
    for (int cc = 0; cc < 8; ++cc) {              // cin chunks of 32
        if (cc + 1 < 8) stage((cc + 1) & 1, cc + 1);   // prefetch next chunk (DMA)
#if USE_ASYNC
        // Newest NPT copies (next chunk) may stay in flight; current chunk done.
        if (cc + 1 < 8) __builtin_amdgcn_s_wait_asynccnt(NPT);
        else            __builtin_amdgcn_s_wait_asynccnt(0);
#endif
        __syncthreads();
        const _Float16* sl = slab[cc & 1];

#pragma unroll
        for (int t = 0; t < 9; ++t) {             // 3x3 taps
            // ---- software pipeline: issue next tap's A loads first ----
            if (!(cc == 7 && t == 8)) {
                int ncc = (t == 8) ? cc + 1 : cc;
                int ntp = (t == 8) ? 0 : t + 1;
                loadA(ncc, ntp, anxt);
            }

            const int ky = t / 3, kx = t % 3;
            v16h bfr[2];
#pragma unroll
            for (int nf = 0; nf < 2; ++nf) {
                // 16-bit B 32x16: lanes0-15 hold K=0..7,16..23; lanes16-31 +8
                int lidx = ((py[nf] - oy0 + ky) * HIN + px[nf] + kx) * 32 + hi * 8;
                union { v16h v; v8h h[2]; } u;
                u.h[0] = *(const v8h*)(sl + lidx);
                u.h[1] = *(const v8h*)(sl + lidx + 16);
                bfr[nf] = u.v;
            }
#pragma unroll
            for (int mf = 0; mf < 2; ++mf)
#pragma unroll
                for (int nf = 0; nf < 2; ++nf)
                    acc[mf][nf] = __builtin_amdgcn_wmma_f32_16x16x32_f16(
                        false, acur[mf], false, bfr[nf],
                        (short)0, acc[mf][nf], false, false);

            acur[0] = anxt[0];
            acur[1] = anxt[1];
        }
        __syncthreads();   // compute done -> buffer may be overwritten next iter
    }

    // ---- epilogue: + beta, store f16.  C/D: M = r + 8*hi, N = l15 ----
#pragma unroll
    for (int mf = 0; mf < 2; ++mf) {
#pragma unroll
        for (int nf = 0; nf < 2; ++nf) {
            int p = pos[nf];
            if (p >= P) continue;
#pragma unroll
            for (int r = 0; r < 8; ++r) {
                int cout = mt * 64 + waveM * 32 + mf * 16 + r + hi * 8;
                Y[((size_t)b * COUT + cout) * P + p] =
                    (_Float16)(acc[mf][nf][r] + beta[cout]);
            }
        }
    }
}

// ---------------------------------------------------------------------------
// Depthwise cross-correlation: z [BC][6][6] (x) x [BC][30][30] -> [BC][25][25]
// grid = NB*COUT blocks, 256 threads
// ---------------------------------------------------------------------------
__global__ __launch_bounds__(256) void xcorr_kernel(
    const _Float16* __restrict__ Z, const _Float16* __restrict__ Xc,
    float* __restrict__ out)
{
    __shared__ float zs[36];
    __shared__ float xs[900];
    const int bc  = blockIdx.x;
    const int tid = threadIdx.x;
    const _Float16* zp = Z  + (size_t)bc * 36;
    const _Float16* xp = Xc + (size_t)bc * 900;
    if (tid < 36) zs[tid] = (float)zp[tid];
    for (int i = tid; i < 900; i += 256) xs[i] = (float)xp[i];
    __syncthreads();
    for (int o = tid; o < 625; o += 256) {
        int oy = o / 25, ox = o % 25;
        float a = 0.f;
#pragma unroll
        for (int ky = 0; ky < 6; ++ky)
#pragma unroll
            for (int kx = 0; kx < 6; ++kx)
                a = fmaf(zs[ky * 6 + kx], xs[(oy + ky) * 30 + ox + kx], a);
        out[(size_t)bc * 625 + o] = a;
    }
}

// ---------------------------------------------------------------------------
extern "C" void kernel_launch(void* const* d_in, const int* in_sizes, int n_in,
                              void* d_out, int out_size, void* d_ws, size_t ws_size,
                              hipStream_t stream) {
    (void)in_sizes; (void)n_in; (void)out_size; (void)ws_size;

    const float* z = (const float*)d_in[0];   // [32,256,8,8]
    const float* x = (const float*)d_in[1];   // [32,256,32,32]

    // Workspace layout (halves unless noted)
    char* ws = (char*)d_ws;
    _Float16* wpk  = (_Float16*)ws;                              // 4 * 256*2304
    size_t    wpkN = (size_t)4 * COUT * KTOT;
    float*    beta = (float*)(ws + wpkN * sizeof(_Float16));     // 4 * 256 f32
    _Float16* z16t = (_Float16*)((char*)beta + 4 * COUT * sizeof(float)); // 32*64*256
    _Float16* x16t = z16t + (size_t)NB * 64 * CIN;               // 32*1024*256
    _Float16* cz   = x16t + (size_t)NB * 1024 * CIN;             // 2*32*256*36
    const size_t czStride = (size_t)NB * COUT * 36;
    _Float16* cx   = cz + 2 * czStride;                          // 2*32*256*900
    const size_t cxStride = (size_t)NB * COUT * 900;

    // 1) fold BN + pack weights (pi=0 cls_z, 1 cls_x, 2 reg_z, 3 reg_x)
    for (int pi = 0; pi < 4; ++pi) {
        const float* w  = (const float*)d_in[2 + pi * 5 + 0];
        const float* g  = (const float*)d_in[2 + pi * 5 + 1];
        const float* bb = (const float*)d_in[2 + pi * 5 + 2];
        const float* m  = (const float*)d_in[2 + pi * 5 + 3];
        const float* v  = (const float*)d_in[2 + pi * 5 + 4];
        prep_weights_kernel<<<KTOT, 256, 0, stream>>>(
            w, g, bb, m, v, wpk + (size_t)pi * COUT * KTOT, beta + pi * COUT);
    }

    // 2) activations -> channel-last f16
    convert_nhwc_kernel<<<NB * 8 * 2, 256, 0, stream>>>(z, z16t, 64);     // HW=64
    convert_nhwc_kernel<<<NB * 8 * 32, 256, 0, stream>>>(x, x16t, 1024);  // HW=1024

    // 3) convs via WMMA implicit GEMM (async double-buffered LDS staging)
    for (int path = 0; path < 2; ++path) {
        int piz = path * 2 + 0, pix = path * 2 + 1;
        conv_gemm_kernel<8><<<NB * 4 * 1, 256, 0, stream>>>(
            z16t, wpk + (size_t)piz * COUT * KTOT, beta + piz * COUT,
            cz + path * czStride, 6, 36, 1);
        conv_gemm_kernel<32><<<NB * 4 * 8, 256, 0, stream>>>(
            x16t, wpk + (size_t)pix * COUT * KTOT, beta + pix * COUT,
            cx + path * cxStride, 30, 900, 8);
    }

    // 4) depthwise xcorr -> f32 outputs, concatenated (cls then reg)
    float* out = (float*)d_out;
    for (int path = 0; path < 2; ++path)
        xcorr_kernel<<<NB * COUT, 256, 0, stream>>>(
            cz + path * czStride, cx + path * cxStride,
            out + (size_t)path * NB * COUT * 625);
}
